// VARNN_12799002542378
// MI455X (gfx1250) — matrix operand order repaired and checked by reading, not compiled, not run on hardware
//
#include <hip/hip_runtime.h>
#include <cstddef>

typedef __bf16 bf16_t;
typedef __attribute__((ext_vector_type(16))) __bf16 v16bf;
typedef __attribute__((ext_vector_type(8)))  __bf16 v8bf;
typedef __attribute__((ext_vector_type(8)))  float   v8f;

// Global-address-space fragment pointer: keeps B loads as global_load_b128
// even after the pointer is laundered through inline asm.
typedef __attribute__((address_space(1))) const v16bf* gfragp;

// Problem constants (match reference)
#define NN 256
#define CC 3
#define TT 300
#define VV 25
#define MM 2
#define HH 256
#define NCLS 60
#define KPAD_IN 96     // 75 padded to 3 k-tiles of 32
#define GATES 1024     // 4*H

__device__ __forceinline__ float sigf(float x) { return 1.0f / (1.0f + __expf(-x)); }

// ---- A fragment load: 16x32 bf16, ISA layout ----
// lanes 0-15 : row = lane,    K = kt+0..7 (v0-3), kt+16..23 (v4-7)
// lanes 16-31: row = lane-16, K = kt+8..15, kt+24..31
// caller passes p already offset by (row*ld + kt*32 + (laneHi?8:0))
__device__ __forceinline__ v16bf load_a_frag(const bf16_t* p) {
    v8bf lo = *(const v8bf*)(p);
    v8bf hi = *(const v8bf*)(p + 16);
    v16bf r;
#pragma unroll
    for (int i = 0; i < 8; ++i) { r[i] = lo[i]; r[8 + i] = hi[i]; }
    return r;
}

// =====================================================================
// Weight pack: W (rows=1024 outputs, cols=Kin) fp32 -> B-fragment bf16.
// Fragment element index: ((ct*nKT + kt)*32 + lane)*16 + e
//   col = ct*16 + (lane&15); k = kt*32 + (e<8 ? (lane>>4)*8 + e : 16 + (lane>>4)*8 + e-8)
// =====================================================================
__global__ void pack_w_kernel(const float* __restrict__ W, bf16_t* __restrict__ Bp,
                              int Kin, int nKT) {
    int idx = blockIdx.x * blockDim.x + threadIdx.x;
    int total = 64 * nKT * 32 * 16;
    if (idx >= total) return;
    int e    = idx & 15;
    int lane = (idx >> 4) & 31;
    int rest = idx >> 9;
    int kt   = rest % nKT;
    int ct   = rest / nKT;
    int col  = ct * 16 + (lane & 15);
    int k    = kt * 32 + ((lane >> 4) << 3) + (e & 7) + ((e >> 3) << 4);
    float v  = (k < Kin) ? W[(size_t)col * Kin + k] : 0.0f;
    Bp[idx]  = (bf16_t)v;
}

__global__ void bias_sum_kernel(const float* __restrict__ bih, const float* __restrict__ bhh,
                                float* __restrict__ bs) {
    int i = blockIdx.x * blockDim.x + threadIdx.x;
    if (i < GATES) bs[i] = bih[i] + bhh[i];
}

// =====================================================================
// seq[n][t][c*25+v] = x[n,c,t,v,m]; cols 75..95 zero. (N,T,96) bf16
// =====================================================================
__global__ void build_seq_kernel(const float* __restrict__ x, bf16_t* __restrict__ seq, int m) {
    int idx = blockIdx.x * blockDim.x + threadIdx.x;
    if (idx >= NN * TT * KPAD_IN) return;
    int col = idx % KPAD_IN;
    int nt  = idx / KPAD_IN;
    int t   = nt % TT;
    int n   = nt / TT;
    float v = 0.0f;
    if (col < CC * VV) {
        int c = col / VV, vv = col % VV;
        v = x[((((size_t)n * CC + c) * TT + t) * VV + vv) * MM + m];
    }
    seq[idx] = (bf16_t)v;
}

// =====================================================================
// Rotation/translation head: per batch row, 3-dim FCs + R matrix build.
// =====================================================================
__global__ void head_kernel(const float* __restrict__ hrot, const float* __restrict__ htr,
                            const float* __restrict__ rotW, const float* __restrict__ rotb,
                            const float* __restrict__ trW,  const float* __restrict__ trb,
                            float* __restrict__ Rm, float* __restrict__ trans) {
    int n = threadIdx.x;
    if (n >= NN) return;
    float a[3], tt[3];
#pragma unroll
    for (int j = 0; j < 3; ++j) {
        float s1 = rotb[j], s2 = trb[j];
        for (int k = 0; k < HH; ++k) {
            float hr = hrot[(size_t)n * HH + k];
            float ht = htr[(size_t)n * HH + k];
            s1 += hr * rotW[j * HH + k];
            s2 += ht * trW[j * HH + k];
        }
        a[j]  = s1 * 3.14159265358979323846f;
        tt[j] = s2;
    }
    float cx = cosf(a[0]), sx = sinf(a[0]);
    float cy = cosf(a[1]), sy = sinf(a[1]);
    float cz = cosf(a[2]), sz = sinf(a[2]);
    float* R = Rm + n * 9;
    R[0] = cy * cz;                R[1] = cy * sz;                R[2] = -sy;
    R[3] = sx * sy * cz - cx * sz; R[4] = sx * sy * sz + cx * cz; R[5] = sx * cy;
    R[6] = cx * sy * cz + sx * sz; R[7] = cx * sy * sz - sx * cz; R[8] = cx * cy;
    trans[n * 3 + 0] = tt[0]; trans[n * 3 + 1] = tt[1]; trans[n * 3 + 2] = tt[2];
}

// =====================================================================
// seq2[n][t][c*25+v] = (R @ (x - trans))[c], bf16, pad to 96
// =====================================================================
__global__ void transform_kernel(const float* __restrict__ x, const float* __restrict__ Rm,
                                 const float* __restrict__ trans, bf16_t* __restrict__ seq2, int m) {
    int idx = blockIdx.x * blockDim.x + threadIdx.x;
    if (idx >= NN * TT) return;
    int t = idx % TT, n = idx / TT;
    const float* R = Rm + n * 9;
    float t0 = trans[n * 3 + 0], t1 = trans[n * 3 + 1], t2 = trans[n * 3 + 2];
    bf16_t* o = seq2 + (size_t)idx * KPAD_IN;
    for (int v = 0; v < VV; ++v) {
        float x0 = x[((((size_t)n * CC + 0) * TT + t) * VV + v) * MM + m] - t0;
        float x1 = x[((((size_t)n * CC + 1) * TT + t) * VV + v) * MM + m] - t1;
        float x2 = x[((((size_t)n * CC + 2) * TT + t) * VV + v) * MM + m] - t2;
        o[v]          = (bf16_t)(R[0] * x0 + R[1] * x1 + R[2] * x2);
        o[VV + v]     = (bf16_t)(R[3] * x0 + R[4] * x1 + R[5] * x2);
        o[2 * VV + v] = (bf16_t)(R[6] * x0 + R[7] * x1 + R[8] * x2);
    }
    for (int c = 75; c < KPAD_IN; ++c) o[c] = (bf16_t)0.0f;
}

// =====================================================================
// Persistent recurrent LSTM kernel (WMMA bf16 -> f32 acc), templated on
// the number of input k-tiles so every A/B address is a compile-time
// immediate offset off a per-wave base pointer. The weight base pointers
// are typed addrspace(1) (-> global_load_b128, no flat/dscnt coupling) and
// laundered through an empty asm each timestep so LICM cannot hoist the
// (t-invariant) fragment loads out of the 300-step loop (no scratch
// spills). Weights stay L2-resident instead.
//
// grid = (#groups)*16 blocks, block = 256 threads (8 wave32).
// Wave w owns col-tiles { g*16 + w, g*16 + w + 8 : g in 0..3 } so each wave
// holds a full (i,f,g,o) gate set in registers -> gate math stays in VGPRs;
// only new bf16 h (8 KB) passes through LDS for the next step's A operand.
// Two independent LSTMs (same input) can share a launch: grp = blockIdx>>4.
// =====================================================================
template <int NKA>
__global__ void __launch_bounds__(256, 1)
lstm_rec_kernel(const bf16_t* __restrict__ Aseq,
                const bf16_t* __restrict__ WihA, const bf16_t* __restrict__ WhhA,
                const float* __restrict__ bsA, bf16_t* __restrict__ outA, float* __restrict__ hlA,
                const bf16_t* __restrict__ WihB, const bf16_t* __restrict__ WhhB,
                const float* __restrict__ bsB, bf16_t* __restrict__ outB, float* __restrict__ hlB,
                int Tn) {
    __shared__ bf16_t hS[16 * HH];   // 16 rows x 256 cols bf16

    const int grp = blockIdx.x >> 4;
    const bf16_t* WihP   = grp ? WihB : WihA;
    const bf16_t* WhhP   = grp ? WhhB : WhhA;
    const float*  bsum   = grp ? bsB  : bsA;
    bf16_t*       outSeq = grp ? outB : outA;
    float*        hLast  = grp ? hlB  : hlA;

    const int tid  = threadIdx.x;
    const int lane = tid & 31;
    const int w    = tid >> 5;          // wave 0..7
    const int lrow = lane & 15;
    const int lhi  = lane >> 4;
    const int rowBase  = (blockIdx.x & 15) * 16;
    const int koffLane = lhi << 3;      // +8 elems for high half lanes
    constexpr int ldA  = NKA * 32;
    constexpr int NK   = NKA + 8;

    for (int i = tid; i < 16 * HH; i += 256) hS[i] = (bf16_t)0.0f;
    __syncthreads();

    // col-tile delta for acc slot s: ct[s] = w + sd[s]; gate = s>>1, part = s&1
    constexpr int sd[8] = {0, 8, 16, 24, 32, 40, 48, 56};
    float bias[8];
#pragma unroll
    for (int s = 0; s < 8; ++s) bias[s] = bsum[(w + sd[s]) * 16 + lrow];

    // per-wave B base pointers in global address space; every fragment is at
    // a constant (immediate) offset. Index unit = v16bf (32B fragments).
    gfragp bihG = (gfragp)(const v16bf*)(WihP + ((size_t)w * NKA * 32 + lane) * 16);
    gfragp bhhG = (gfragp)(const v16bf*)(WhhP + ((size_t)w * 8 * 32 + lane) * 16);
    const bf16_t* hrow = &hS[lrow * HH + koffLane];

    v8f c0 = {}; v8f c1 = {};

    for (int t = 0; t < Tn; ++t) {
        // Launder weight bases: redefined every iteration -> fragment loads
        // cannot be hoisted out of the t loop (no spill); offsets stay imm;
        // addrspace(1) type keeps them as global_load_b128.
        gfragp bih_t = bihG;
        gfragp bhh_t = bhhG;
        asm volatile("" : "+v"(bih_t), "+v"(bhh_t));

        v8f acc[8];
#pragma unroll
        for (int s = 0; s < 8; ++s) {
            v8f z;
#pragma unroll
            for (int i = 0; i < 8; ++i) z[i] = bias[s];
            acc[s] = z;
        }

        const bf16_t* arow = Aseq + ((size_t)(rowBase + lrow) * Tn + t) * ldA + koffLane;
        if (t + 1 < Tn) __builtin_prefetch(arow + ldA, 0, 3);   // warm next step's A row

        // ---- k loop: x part (global A) then h part (LDS A); per k-tile:
        // clause of A + 8 B fragment loads, then 8 back-to-back WMMAs.
#pragma unroll
        for (int kt = 0; kt < NK; ++kt) {
            v16bf a;
            v16bf b[8];
            if (kt < NKA) {
                a = load_a_frag(arow + kt * 32);
#pragma unroll
                for (int s = 0; s < 8; ++s)
                    b[s] = bih_t[(size_t)(sd[s] * NKA + kt) * 32];
            } else {
                a = load_a_frag(hrow + (kt - NKA) * 32);
#pragma unroll
                for (int s = 0; s < 8; ++s)
                    b[s] = bhh_t[(size_t)(sd[s] * 8 + (kt - NKA)) * 32];
            }
#pragma unroll
            for (int s = 0; s < 8; ++s)
                acc[s] = __builtin_amdgcn_wmma_f32_16x16x32_bf16(
                    false, a, false, b[s], (short)0, acc[s], false, false);
        }
        __syncthreads();   // all reads of hS complete before rewrite

        // ---- gate math fully in registers; write new h ----
#pragma unroll
        for (int p = 0; p < 2; ++p) {
            v8f* cp = p ? &c1 : &c0;
#pragma unroll
            for (int i = 0; i < 8; ++i) {
                float ci = sigf(acc[2 + p][i]) * (*cp)[i] +
                           sigf(acc[0 + p][i]) * tanhf(acc[4 + p][i]);
                (*cp)[i] = ci;
                float hv = sigf(acc[6 + p][i]) * tanhf(ci);
                int row = i + 8 * lhi;
                int col = (w + 8 * p) * 16 + lrow;
                hS[row * HH + col] = (bf16_t)hv;
                if (outSeq)
                    outSeq[((size_t)(rowBase + row) * Tn + t) * HH + col] = (bf16_t)hv;
                if (hLast && t == Tn - 1)
                    hLast[(size_t)(rowBase + row) * HH + col] = hv;
            }
        }
        __syncthreads();   // h visible before next step's reads
    }
}

// =====================================================================
// out[n][cls] = fcb[cls] + sum_k max(f0,f1)[n][k] * fcW[cls][k]
// =====================================================================
__global__ void final_kernel(const float* __restrict__ f0, const float* __restrict__ f1,
                             const float* __restrict__ W, const float* __restrict__ b,
                             float* __restrict__ out) {
    int idx = blockIdx.x * blockDim.x + threadIdx.x;
    if (idx >= NN * NCLS) return;
    int cls = idx % NCLS, n = idx / NCLS;
    float s = b[cls];
    for (int k = 0; k < HH; ++k)
        s += fmaxf(f0[(size_t)n * HH + k], f1[(size_t)n * HH + k]) * W[(size_t)cls * HH + k];
    out[idx] = s;
}

// =====================================================================
extern "C" void kernel_launch(void* const* d_in, const int* in_sizes, int n_in,
                              void* d_out, int out_size, void* d_ws, size_t ws_size,
                              hipStream_t stream) {
    (void)in_sizes; (void)n_in; (void)out_size; (void)ws_size;
    const float* x       = (const float*)d_in[0];
    const float* rot_Wih = (const float*)d_in[1];
    const float* rot_Whh = (const float*)d_in[2];
    const float* rot_bih = (const float*)d_in[3];
    const float* rot_bhh = (const float*)d_in[4];
    const float* rot_fcW = (const float*)d_in[5];
    const float* rot_fcb = (const float*)d_in[6];
    const float* tr_Wih  = (const float*)d_in[7];
    const float* tr_Whh  = (const float*)d_in[8];
    const float* tr_bih  = (const float*)d_in[9];
    const float* tr_bhh  = (const float*)d_in[10];
    const float* tr_fcW  = (const float*)d_in[11];
    const float* tr_fcb  = (const float*)d_in[12];
    const float* m_Wih[3] = {(const float*)d_in[13], (const float*)d_in[17], (const float*)d_in[21]};
    const float* m_Whh[3] = {(const float*)d_in[14], (const float*)d_in[18], (const float*)d_in[22]};
    const float* m_bih[3] = {(const float*)d_in[15], (const float*)d_in[19], (const float*)d_in[23]};
    const float* m_bhh[3] = {(const float*)d_in[16], (const float*)d_in[20], (const float*)d_in[24]};
    const float* fcW = (const float*)d_in[25];
    const float* fcb = (const float*)d_in[26];
    float* out = (float*)d_out;

    // ---- workspace carve-out (256B aligned) ----
    char* ws = (char*)d_ws;
    size_t off = 0;
    auto carve = [&](size_t bytes) -> char* {
        char* p = ws + off;
        off += (bytes + 255) & ~(size_t)255;
        return p;
    };
    bf16_t* seq   = (bf16_t*)carve((size_t)NN * TT * KPAD_IN * 2);
    bf16_t* seq2  = (bf16_t*)carve((size_t)NN * TT * KPAD_IN * 2);
    bf16_t* hseqA = (bf16_t*)carve((size_t)NN * TT * HH * 2);
    bf16_t* hseqB = (bf16_t*)carve((size_t)NN * TT * HH * 2);
    const size_t szWih96  = (size_t)64 * 3 * 32 * 16;  // elements
    const size_t szW256   = (size_t)64 * 8 * 32 * 16;
    bf16_t* rotWihP = (bf16_t*)carve(szWih96 * 2);
    bf16_t* rotWhhP = (bf16_t*)carve(szW256 * 2);
    bf16_t* trWihP  = (bf16_t*)carve(szWih96 * 2);
    bf16_t* trWhhP  = (bf16_t*)carve(szW256 * 2);
    bf16_t* mWihP[3]; bf16_t* mWhhP[3];
    mWihP[0] = (bf16_t*)carve(szWih96 * 2);
    mWihP[1] = (bf16_t*)carve(szW256 * 2);
    mWihP[2] = (bf16_t*)carve(szW256 * 2);
    for (int l = 0; l < 3; ++l) mWhhP[l] = (bf16_t*)carve(szW256 * 2);
    float* bs_rot = (float*)carve(GATES * 4);
    float* bs_tr  = (float*)carve(GATES * 4);
    float* bs_m[3];
    for (int l = 0; l < 3; ++l) bs_m[l] = (float*)carve(GATES * 4);
    float* h_rot = (float*)carve((size_t)NN * HH * 4);
    float* h_tr  = (float*)carve((size_t)NN * HH * 4);
    float* Rbuf  = (float*)carve((size_t)NN * 9 * 4);
    float* tbuf  = (float*)carve((size_t)NN * 3 * 4);
    float* feat[2];
    feat[0] = (float*)carve((size_t)NN * HH * 4);
    feat[1] = (float*)carve((size_t)NN * HH * 4);

    // ---- pack weights + bias sums (once per call; deterministic) ----
    auto packW = [&](const float* W, bf16_t* Bp, int Kin, int nKT) {
        int total = 64 * nKT * 512;
        pack_w_kernel<<<(total + 255) / 256, 256, 0, stream>>>(W, Bp, Kin, nKT);
    };
    packW(rot_Wih, rotWihP, CC * VV, 3);
    packW(rot_Whh, rotWhhP, HH, 8);
    packW(tr_Wih,  trWihP,  CC * VV, 3);
    packW(tr_Whh,  trWhhP,  HH, 8);
    packW(m_Wih[0], mWihP[0], CC * VV, 3);
    packW(m_Wih[1], mWihP[1], HH, 8);
    packW(m_Wih[2], mWihP[2], HH, 8);
    for (int l = 0; l < 3; ++l) packW(m_Whh[l], mWhhP[l], HH, 8);
    bias_sum_kernel<<<4, 256, 0, stream>>>(rot_bih, rot_bhh, bs_rot);
    bias_sum_kernel<<<4, 256, 0, stream>>>(tr_bih, tr_bhh, bs_tr);
    for (int l = 0; l < 3; ++l)
        bias_sum_kernel<<<4, 256, 0, stream>>>(m_bih[l], m_bhh[l], bs_m[l]);

    const int seqThreads = NN * TT * KPAD_IN;
    for (int m = 0; m < 2; ++m) {
        build_seq_kernel<<<(seqThreads + 255) / 256, 256, 0, stream>>>(x, seq, m);
        // rot + tr view LSTMs fused in one launch (independent, same input)
        lstm_rec_kernel<3><<<32, 256, 0, stream>>>(
            seq,
            rotWihP, rotWhhP, bs_rot, nullptr, h_rot,
            trWihP,  trWhhP,  bs_tr,  nullptr, h_tr, TT);
        head_kernel<<<1, 256, 0, stream>>>(h_rot, h_tr, rot_fcW, rot_fcb, tr_fcW, tr_fcb,
                                           Rbuf, tbuf);
        transform_kernel<<<(NN * TT + 255) / 256, 256, 0, stream>>>(x, Rbuf, tbuf, seq2, m);
        // stacked main LSTM (dependent chain)
        lstm_rec_kernel<3><<<16, 256, 0, stream>>>(
            seq2,
            mWihP[0], mWhhP[0], bs_m[0], hseqA, nullptr,
            mWihP[0], mWhhP[0], bs_m[0], hseqA, nullptr, TT);
        lstm_rec_kernel<8><<<16, 256, 0, stream>>>(
            hseqA,
            mWihP[1], mWhhP[1], bs_m[1], hseqB, nullptr,
            mWihP[1], mWhhP[1], bs_m[1], hseqB, nullptr, TT);
        lstm_rec_kernel<8><<<16, 256, 0, stream>>>(
            hseqB,
            mWihP[2], mWhhP[2], bs_m[2], nullptr, feat[m],
            mWihP[2], mWhhP[2], bs_m[2], nullptr, feat[m], TT);
    }
    final_kernel<<<(NN * NCLS + 255) / 256, 256, 0, stream>>>(feat[0], feat[1], fcW, fcb, out);
}
